// ReplicatedLinearWithLoRA_79800492359938
// MI455X (gfx1250) — matrix operand order repaired and checked by reading, not compiled, not run on hardware
//
#include <hip/hip_runtime.h>
#include <hip/hip_bf16.h>
#include <stdint.h>

// MI455X / gfx1250: wave32, WMMA 16x16x32 bf16.
// out = x @ W_eff^T + bias, W_eff = W + LoRA(B@A) folded in (prepass).
// Compute-bound op (~50us WMMA vs ~16us HBM) -> convert x to bf16 ONCE,
// feed GEMM purely via async global->LDS DMA, depth-3 pipelined.

typedef __attribute__((ext_vector_type(16))) __bf16 v16bf;
typedef __attribute__((ext_vector_type(8)))  float  v8f;

#define S_TOK  8192
#define H_DIM  7168
#define D0_DIM 1536
#define D1_DIM 576
#define DT_DIM 2112      // D0 + D1
#define R_LORA 64

#define BM  128
#define BN  64
#define BK  64
#define LDK 72           // padded LDS row stride in bf16 elements (144 B)
#define NKT (H_DIM / BK) // 112

// ---- CDNA5 feature gates (device pass only; host pass takes fallbacks) ----
#if defined(__AMDGCN__) && __has_builtin(__builtin_amdgcn_global_load_async_to_lds_b128) && \
    __has_builtin(__builtin_amdgcn_s_wait_asynccnt)
#define USE_ASYNC_LDS 1
#else
#define USE_ASYNC_LDS 0
#endif

#if defined(__AMDGCN__) && __has_builtin(__builtin_amdgcn_cvt_pk_bf16_f32)
#define USE_PK_CVT 1
#else
#define USE_PK_CVT 0
#endif

static __device__ __forceinline__ unsigned short f2bf(float f) {
    union { float f; unsigned u; } v; v.f = f;
    unsigned r = v.u + 0x7FFFu + ((v.u >> 16) & 1u);  // round-to-nearest-even
    return (unsigned short)(r >> 16);
}

// pack two fp32 -> one dword of two bf16
static __device__ __forceinline__ unsigned pack2bf(float a, float b) {
#if USE_PK_CVT
    typedef __attribute__((ext_vector_type(2))) __bf16 v2bf;
    union { v2bf v; unsigned u; } u;
    u.v = __builtin_amdgcn_cvt_pk_bf16_f32(a, b);
    return u.u;
#else
    // plain casts: let the backend pick a native cvt if available
    typedef __attribute__((ext_vector_type(2))) __bf16 v2bf;
    union { v2bf v; unsigned u; } u;
    u.v[0] = (__bf16)a; u.v[1] = (__bf16)b;
    return u.u;
#endif
}

// global -> LDS, 16B per lane (async DMA on CDNA5, plain copy otherwise)
#if USE_ASYNC_LDS
typedef int v4i_vs __attribute__((vector_size(16)));
typedef __attribute__((address_space(1))) v4i_vs gv4i_t;
typedef __attribute__((address_space(3))) v4i_vs lv4i_t;
#endif
static __device__ __forceinline__ void g2l_b128(const void* g, void* l) {
#if USE_ASYNC_LDS
    __builtin_amdgcn_global_load_async_to_lds_b128(
        (gv4i_t*)(unsigned long long)(uintptr_t)g,
        (lv4i_t*)(unsigned)(uintptr_t)l,
        0, 0);                                   // ASYNCcnt++
#else
    *(uint4*)l = *(const uint4*)g;
#endif
}

// ---------------------------------------------------------------------------
// Prepass 1: W_eff[o][h] = W[o][h] + sum_r B[slice][o'][r] * A[slice*R+r][h]
// ---------------------------------------------------------------------------
__global__ __launch_bounds__(256) void build_weff_kernel(
    const float* __restrict__ W, const float* __restrict__ A,
    const float* __restrict__ B, unsigned short* __restrict__ Weff)
{
    const int o     = blockIdx.x;
    const int slice = (o >= D0_DIM) ? 1 : 0;
    const int orow  = slice ? (o - D0_DIM) : o;
    const float* Brow  = B + ((size_t)slice * D0_DIM + orow) * R_LORA;
    const float* Arows = A + (size_t)slice * R_LORA * H_DIM;

    __shared__ float sB[R_LORA];
    if (threadIdx.x < R_LORA) sB[threadIdx.x] = Brow[threadIdx.x];
    __syncthreads();

    for (int h = threadIdx.x; h < H_DIM; h += 256) {
        float acc = W[(size_t)o * H_DIM + h];
        #pragma unroll 16
        for (int r = 0; r < R_LORA; ++r)
            acc = fmaf(sB[r], Arows[(size_t)r * H_DIM + h], acc);
        Weff[(size_t)o * H_DIM + h] = f2bf(acc);
    }
}

// ---------------------------------------------------------------------------
// Prepass 2: x fp32 -> bf16 (once; kills the 33x in-loop conversion cost)
// 8 elements per thread, float4 loads -> uint4 store.
// ---------------------------------------------------------------------------
__global__ __launch_bounds__(256) void xcvt_kernel(
    const float* __restrict__ x, unsigned short* __restrict__ xb)
{
    const size_t i = ((size_t)blockIdx.x * 256 + threadIdx.x) * 8;
    float4 a = *(const float4*)(x + i);
    float4 b = *(const float4*)(x + i + 4);
    uint4 pk;
    pk.x = pack2bf(a.x, a.y); pk.y = pack2bf(a.z, a.w);
    pk.z = pack2bf(b.x, b.y); pk.w = pack2bf(b.z, b.w);
    *(uint4*)(xb + i) = pk;
}

// ---------------------------------------------------------------------------
// Main GEMM: out[S,DT] = x @ Weff^T + bias.
// Block 128x64, 8 waves (4Mx2N), wave 32x32 via 2x2 WMMA accumulators.
// PRECONV=1: x already bf16; both tiles async-DMA'd, 4 LDS buffers, depth-3.
// PRECONV=0: fallback, x fp32 converted in registers, double buffered.
// ---------------------------------------------------------------------------
template <bool PRECONV>
__global__ __launch_bounds__(256) void gemm_wmma_kernel(
    const void* __restrict__ xsrc, const unsigned short* __restrict__ Weff,
    const float* __restrict__ bias, float* __restrict__ out)
{
    constexpr int NB    = PRECONV ? 4 : 2;
    constexpr int DEPTH = PRECONV ? 3 : 1;

    __shared__ __align__(16) unsigned short lx[NB][BM][LDK];
    __shared__ __align__(16) unsigned short lw[NB][BN][LDK];

    const int tid  = threadIdx.x;
    const int lane = tid & 31;
    const int wave = tid >> 5;
    const int wm   = wave & 3;   // wave M index -> 32-row strip
    const int wn   = wave >> 2;  // wave N index -> 32-col strip
    const int m0   = blockIdx.y * BM;
    const int n0   = blockIdx.x * BN;

    v8f acc[2][2];
    #pragma unroll
    for (int mt = 0; mt < 2; ++mt)
        #pragma unroll
        for (int nt = 0; nt < 2; ++nt)
            #pragma unroll
            for (int j = 0; j < 8; ++j)
                acc[mt][nt][j] = 0.0f;

    // 16B-chunk loader mappings (8 chunks/row, 8 threads/row, 32 rows/pass)
    const int cr_row = tid >> 3;
    const int cr_col = (tid & 7) * 8;

    auto compute = [&](int buf) {
        #pragma unroll
        for (int kk = 0; kk < 2; ++kk) {                       // two K=32 steps
            union Frag { uint4 q[2]; v16bf v; };
            Frag afrag[2], bfrag[2];

            // A fragment (ISA 16-bit A 16x32 layout):
            // lanes 0-15 : row = lane,    K = {0..7, 16..23}
            // lanes 16-31: row = lane-16, K = {8..15, 24..31}
            const int arow = wm * 32 + (lane & 15);
            const int akb  = kk * 32 + ((lane >> 4) ? 8 : 0);
            #pragma unroll
            for (int mt = 0; mt < 2; ++mt) {
                const unsigned short* ap = &lx[buf][arow + mt * 16][akb];
                afrag[mt].q[0] = *(const uint4*)(ap);
                afrag[mt].q[1] = *(const uint4*)(ap + 16);
            }

            // B fragment: lane = column N, K = {0..7,16..23} / {8..15,24..31} per half-wave.
            // B[k][n] = Weff[n][k] -> Weff rows are B columns.
            const int brow = wn * 32 + (lane & 15);
            const int bkb  = kk * 32 + ((lane >> 4) ? 8 : 0);
            #pragma unroll
            for (int nt = 0; nt < 2; ++nt) {
                const unsigned short* bp = &lw[buf][brow + nt * 16][bkb];
                bfrag[nt].q[0] = *(const uint4*)(bp);
                bfrag[nt].q[1] = *(const uint4*)(bp + 16);
            }

            #pragma unroll
            for (int mt = 0; mt < 2; ++mt)
                #pragma unroll
                for (int nt = 0; nt < 2; ++nt)
                    acc[mt][nt] = __builtin_amdgcn_wmma_f32_16x16x32_bf16(
                        false, afrag[mt].v, false, bfrag[nt].v,
                        (short)0, acc[mt][nt], false, false);
        }
    };

    if constexpr (PRECONV) {
        const unsigned short* xb = (const unsigned short*)xsrc;

        // one tile-set = 6 async b128 instructions per wave (4 X + 2 W)
        auto issue_set = [&](int kt, int buf) {
            const int k0 = kt * BK;
            #pragma unroll
            for (int p = 0; p < 4; ++p) {
                const int r = cr_row + p * 32;
                g2l_b128(xb + (size_t)(m0 + r) * H_DIM + k0 + cr_col,
                         &lx[buf][r][cr_col]);
            }
            #pragma unroll
            for (int p = 0; p < 2; ++p) {
                const int r = cr_row + p * 32;
                g2l_b128(Weff + (size_t)(n0 + r) * H_DIM + k0 + cr_col,
                         &lw[buf][r][cr_col]);
            }
        };

        issue_set(0, 0);
        issue_set(1, 1);
        issue_set(2, 2);

        for (int kt = 0; kt < NKT; ++kt) {
#if USE_ASYNC_LDS
            // async loads retire in order: <=12 left => oldest set (kt) done
            if (kt + 2 < NKT)      __builtin_amdgcn_s_wait_asynccnt(12);
            else if (kt + 1 < NKT) __builtin_amdgcn_s_wait_asynccnt(6);
            else                   __builtin_amdgcn_s_wait_asynccnt(0);
#endif
            __syncthreads();   // publish set kt; all waves done with kt-1
            if (kt + DEPTH < NKT) issue_set(kt + DEPTH, (kt + DEPTH) & 3);
            compute(kt & 3);
        }
    } else {
        const float* xf = (const float*)xsrc;
        const int xr_row = tid >> 4;          // 16 rows/pass, 8 passes
        const int xr_col = (tid & 15) * 4;    // float4 per thread
        float4 xreg[8];

        auto load_global = [&](int kt, int buf) {
            const int k0 = kt * BK;
            #pragma unroll
            for (int p = 0; p < 8; ++p) {
                const int r = xr_row + p * 16;
                xreg[p] = *(const float4*)(xf + (size_t)(m0 + r) * H_DIM + k0 + xr_col);
            }
            #pragma unroll
            for (int p = 0; p < 2; ++p) {
                const int r = cr_row + p * 32;
                g2l_b128(Weff + (size_t)(n0 + r) * H_DIM + k0 + cr_col,
                         &lw[buf][r][cr_col]);
            }
        };
        auto store_lds = [&](int buf) {
            #pragma unroll
            for (int p = 0; p < 8; ++p) {
                const int r = xr_row + p * 16;
                uint2 pk;
                pk.x = pack2bf(xreg[p].x, xreg[p].y);
                pk.y = pack2bf(xreg[p].z, xreg[p].w);
                *(uint2*)&lx[buf][r][xr_col] = pk;
            }
        };

        load_global(0, 0);
        store_lds(0);
#if USE_ASYNC_LDS
        __builtin_amdgcn_s_wait_asynccnt(0);
#endif
        __syncthreads();

        for (int kt = 0; kt < NKT; ++kt) {
            const int cur = kt & 1;
            if (kt + 1 < NKT) load_global(kt + 1, cur ^ 1);
            compute(cur);
            if (kt + 1 < NKT) store_lds(cur ^ 1);
#if USE_ASYNC_LDS
            __builtin_amdgcn_s_wait_asynccnt(0);
#endif
            __syncthreads();
        }
    }

    // epilogue: C layout VGPR j -> M = j + 8*(lane>=16), N = lane&15
    #pragma unroll
    for (int mt = 0; mt < 2; ++mt) {
        #pragma unroll
        for (int nt = 0; nt < 2; ++nt) {
            const int col  = n0 + wn * 32 + nt * 16 + (lane & 15);
            const int row0 = m0 + wm * 32 + mt * 16 + ((lane >> 4) << 3);
            const float bv = bias[col];
            #pragma unroll
            for (int j = 0; j < 8; ++j)
                out[(size_t)(row0 + j) * DT_DIM + col] = acc[mt][nt][j] + bv;
        }
    }
}

extern "C" void kernel_launch(void* const* d_in, const int* in_sizes, int n_in,
                              void* d_out, int out_size, void* d_ws, size_t ws_size,
                              hipStream_t stream) {
    (void)in_sizes; (void)n_in; (void)out_size;
    const float* x    = (const float*)d_in[0];
    const float* W    = (const float*)d_in[1];
    const float* bias = (const float*)d_in[2];
    const float* A    = (const float*)d_in[3];
    const float* B    = (const float*)d_in[4];
    float* out = (float*)d_out;

    const size_t weff_bytes = (size_t)DT_DIM * H_DIM * 2;   // ~30.3 MB
    const size_t xbf_bytes  = (size_t)S_TOK * H_DIM * 2;    // ~117.4 MB

    unsigned short* Weff = (unsigned short*)d_ws;
    build_weff_kernel<<<DT_DIM, 256, 0, stream>>>(W, A, B, Weff);

    dim3 grid(DT_DIM / BN, S_TOK / BM);  // (33, 64)

    if (ws_size >= weff_bytes + xbf_bytes) {
        // preconverted-x fast path: zero conversion VALU in the GEMM loop
        unsigned short* xb = (unsigned short*)((char*)d_ws + weff_bytes);
        xcvt_kernel<<<(S_TOK * (size_t)H_DIM) / (256 * 8), 256, 0, stream>>>(x, xb);
        gemm_wmma_kernel<true><<<grid, 256, 0, stream>>>(xb, Weff, bias, out);
    } else {
        gemm_wmma_kernel<false><<<grid, 256, 0, stream>>>(x, Weff, bias, out);
    }
}
